// TransformerEncoderBlock_88399016886858
// MI455X (gfx1250) — compile-verified, matrix operand order checked by reading, not run on hardware
//
#include <hip/hip_runtime.h>
#include <hip/hip_bf16.h>
#include <math.h>

typedef __attribute__((ext_vector_type(16))) __bf16 v16bf;
typedef __attribute__((ext_vector_type(8)))  float  v8f;
typedef __attribute__((ext_vector_type(4)))  unsigned int u32x4;
typedef __attribute__((ext_vector_type(8)))  int i32x8;
typedef __attribute__((ext_vector_type(4)))  int i32x4;

#define D_MODEL 1024
#define NHEAD   16
#define DHEAD   64
#define DIM_FF  4096
#define BATCH   4
#define SEQ     2048
#define MROWS   (BATCH * SEQ)   // 8192

enum { EP_QKV = 0, EP_RESID = 1, EP_BIAS_GELU = 2, EP_BIAS_RESID = 3 };

#if __has_builtin(__builtin_amdgcn_tensor_load_to_lds)
#define HAVE_TDM 1
#else
#define HAVE_TDM 0
#endif

// ---------------------------------------------------------------------------
// fp32 -> bf16 convert (vectorized x4)
// ---------------------------------------------------------------------------
__global__ __launch_bounds__(256)
void cvt_f32_bf16(const float* __restrict__ in, __bf16* __restrict__ out, int n) {
    int i = (blockIdx.x * 256 + threadIdx.x) * 4;
    if (i < n) {
        float4 v = *(const float4*)(in + i);
        out[i + 0] = (__bf16)v.x;
        out[i + 1] = (__bf16)v.y;
        out[i + 2] = (__bf16)v.z;
        out[i + 3] = (__bf16)v.w;
    }
}

// ---------------------------------------------------------------------------
// LayerNorm over D=1024, writes bf16.  One block (8 waves) per row.
// ---------------------------------------------------------------------------
__global__ __launch_bounds__(256)
void ln_to_bf16(const float* __restrict__ x, const float* __restrict__ sc,
                const float* __restrict__ bi, __bf16* __restrict__ out) {
    const int row = blockIdx.x;
    const float* xr = x + (size_t)row * D_MODEL;
    const int t = threadIdx.x, lane = t & 31, wid = t >> 5;

    float v[4]; float s = 0.f;
    #pragma unroll
    for (int i = 0; i < 4; ++i) { v[i] = xr[t + 256 * i]; s += v[i]; }
    #pragma unroll
    for (int m = 16; m >= 1; m >>= 1) s += __shfl_xor(s, m, 32);

    __shared__ float red[8];
    if (lane == 0) red[wid] = s;
    __syncthreads();
    float mu = 0.f;
    #pragma unroll
    for (int i = 0; i < 8; ++i) mu += red[i];
    mu *= (1.f / D_MODEL);

    float vs = 0.f;
    #pragma unroll
    for (int i = 0; i < 4; ++i) { float d = v[i] - mu; vs += d * d; }
    #pragma unroll
    for (int m = 16; m >= 1; m >>= 1) vs += __shfl_xor(vs, m, 32);
    __syncthreads();
    if (lane == 0) red[wid] = vs;
    __syncthreads();
    float var = 0.f;
    #pragma unroll
    for (int i = 0; i < 8; ++i) var += red[i];
    var *= (1.f / D_MODEL);
    const float rstd = rsqrtf(var + 1e-5f);

    #pragma unroll
    for (int i = 0; i < 4; ++i) {
        int c = t + 256 * i;
        out[(size_t)row * D_MODEL + c] = (__bf16)((v[i] - mu) * rstd * sc[c] + bi[c]);
    }
}

// ---------------------------------------------------------------------------
// TDM descriptor helpers (CDNA5 ISA ch.8, D# groups).
// 2D bf16 tile: tile_dim0 = 32 elems (64B row), tile_dim1 = 128 rows,
// LDS padding 8 DWORDs (32B) every 16 DWORDs (64B) -> 96B (=48 elem) LDS row.
// ---------------------------------------------------------------------------
#if HAVE_TDM
__device__ __forceinline__ u32x4 tdm_group0(unsigned int lds_addr, const void* gptr) {
    const unsigned long long ga = (unsigned long long)(uintptr_t)gptr;
    u32x4 g0;
    g0[0] = 1u;                                            // count=1, user mode
    g0[1] = lds_addr;                                      // LDS byte address
    g0[2] = (unsigned int)ga;                              // global_addr[31:0]
    g0[3] = ((unsigned int)(ga >> 32) & 0x01FFFFFFu)       // global_addr[56:32]
            | (2u << 30);                                  // type=2 ("image")
    return g0;
}

__device__ __forceinline__ i32x8 tdm_group1(int K, int dim1) {
    i32x8 g1;
    g1[0] = (1 << 16)      // data_size = 1 -> 2 bytes
          | (1 << 20)      // pad_enable
          | (3 << 22)      // pad_interval: 16 DWORDs (64B) before pad
          | (7 << 25);     // pad_amount: 8 DWORDs (32B)
    g1[1] = (K & 0xFFFF) << 16;                            // tensor_dim0[15:0]
    g1[2] = ((K >> 16) & 0xFFFF) | ((dim1 & 0xFFFF) << 16);// dim0 hi | dim1 lo
    g1[3] = ((dim1 >> 16) & 0xFFFF) | (32 << 16);          // dim1 hi | tile_dim0=32
    g1[4] = 128;                                           // tile_dim1=128, tile_dim2=0
    g1[5] = K;                                             // tensor_dim0_stride[31:0]
    g1[6] = 0;                                             // stride hi | dim1_stride lo
    g1[7] = 0;
    return g1;
}

__device__ __forceinline__ void tdm_load(u32x4 g0, i32x8 g1) {
    const i32x4 z4 = {0, 0, 0, 0};
#if defined(__clang_major__) && (__clang_major__ >= 23)
    const i32x8 z8 = {0, 0, 0, 0, 0, 0, 0, 0};
    __builtin_amdgcn_tensor_load_to_lds(g0, g1, z4, z4, z8, 0);
#else
    __builtin_amdgcn_tensor_load_to_lds(g0, g1, z4, z4, 0);
#endif
}
#endif

// ---------------------------------------------------------------------------
// Tiled bf16 GEMM:  C[M,N] = A[M,K] * W[N,K]^T   (torch Linear layout)
// 128x128 block tile, 8 waves, each wave 32x64 (2x4 WMMA 16x16 tiles),
// k-step 32.  Tile staging via Tensor Data Mover (wave 0 issues the DMA,
// gates with s_wait_tensorcnt + barrier); VMEM fallback if no TDM builtin.
// ---------------------------------------------------------------------------
template<int EPI>
__global__ __launch_bounds__(256)
void gemm_bf16(const __bf16* __restrict__ A, const __bf16* __restrict__ W,
               int M, int N, int K,
               const float* __restrict__ resid, const float* __restrict__ bias,
               float* __restrict__ outF, __bf16* __restrict__ outB,
               __bf16* __restrict__ q_out, __bf16* __restrict__ k_out,
               __bf16* __restrict__ v_out) {
    __shared__ __attribute__((aligned(32))) __bf16 As[128 * 48];
    __shared__ __attribute__((aligned(32))) __bf16 Bs[128 * 48];

    const int tid = threadIdx.x;
    const int lane = tid & 31, wid = tid >> 5;
    const int waveM = wid & 3, waveN = wid >> 2;          // 4 x 2 wave grid
    const int bm = blockIdx.y * 128, bn = blockIdx.x * 128;
    const int fr = lane & 15;                             // fragment row
    const int halfSel = lane >> 4;                        // contiguous K-half: k = 16*halfSel + s

    const v8f vzero = {0.f,0.f,0.f,0.f,0.f,0.f,0.f,0.f};
    v8f acc[2][4];
    #pragma unroll
    for (int mi = 0; mi < 2; ++mi)
        #pragma unroll
        for (int ni = 0; ni < 4; ++ni) acc[mi][ni] = vzero;

#if HAVE_TDM
    // flat address low 32 bits of an LDS object == LDS byte offset
    const unsigned int ldsA = (unsigned int)(uintptr_t)&As[0];
    const unsigned int ldsB = (unsigned int)(uintptr_t)&Bs[0];
    const i32x8 g1a = tdm_group1(K, M);
    const i32x8 g1b = tdm_group1(K, N);
#else
    const int lrow = tid >> 1;                            // 0..127
    const int lhalf = tid & 1;                            // 16-col half
#endif

    for (int kt = 0; kt < K; kt += 32) {
        __syncthreads();
#if HAVE_TDM
        if (wid == 0) {
            tdm_load(tdm_group0(ldsA, &A[(size_t)bm * K + kt]), g1a);
            tdm_load(tdm_group0(ldsB, &W[(size_t)bn * K + kt]), g1b);
            __builtin_amdgcn_s_wait_tensorcnt(0);
        }
#else
        *(v16bf*)&As[lrow * 48 + lhalf * 16] =
            *(const v16bf*)&A[(size_t)(bm + lrow) * K + kt + lhalf * 16];
        *(v16bf*)&Bs[lrow * 48 + lhalf * 16] =
            *(const v16bf*)&W[(size_t)(bn + lrow) * K + kt + lhalf * 16];
#endif
        __syncthreads();

        v16bf af[2], bf[4];
        #pragma unroll
        for (int mi = 0; mi < 2; ++mi)
            af[mi] = *(const v16bf*)&As[(waveM * 32 + mi * 16 + fr) * 48 + halfSel * 16];
        #pragma unroll
        for (int ni = 0; ni < 4; ++ni)
            bf[ni] = *(const v16bf*)&Bs[(waveN * 64 + ni * 16 + fr) * 48 + halfSel * 16];

        #pragma unroll
        for (int mi = 0; mi < 2; ++mi)
            #pragma unroll
            for (int ni = 0; ni < 4; ++ni)
                acc[mi][ni] = __builtin_amdgcn_wmma_f32_16x16x32_bf16(
                    false, af[mi], false, bf[ni], (short)0, acc[mi][ni], false, false);
    }

    // ---- epilogue (C layout: vgpr r -> row r (lanes 0-15) / row 8+r (16-31)) ----
    #pragma unroll
    for (int mi = 0; mi < 2; ++mi) {
        const int mbase = bm + waveM * 32 + mi * 16 + halfSel * 8;
        #pragma unroll
        for (int ni = 0; ni < 4; ++ni) {
            const int n = bn + waveN * 64 + ni * 16 + fr;
            #pragma unroll
            for (int r = 0; r < 8; ++r) {
                const int m = mbase + r;
                const float c = acc[mi][ni][r];
                if constexpr (EPI == EP_QKV) {
                    const int b = m >> 11, t = m & (SEQ - 1);
                    const int type = n >> 10, dall = n & (D_MODEL - 1);
                    const int h = dall >> 6, dh = dall & 63;
                    const size_t idx =
                        (((size_t)(b * NHEAD + h) * SEQ) + t) * DHEAD + dh;
                    __bf16* dst = (type == 0) ? q_out : (type == 1) ? k_out : v_out;
                    dst[idx] = (__bf16)c;
                } else if constexpr (EPI == EP_RESID) {
                    outF[(size_t)m * N + n] = resid[(size_t)m * N + n] + c;
                } else if constexpr (EPI == EP_BIAS_GELU) {
                    const float xg = c + bias[n];
                    const float g = 0.5f * xg * (1.0f + erff(xg * 0.70710678118f));
                    outB[(size_t)m * N + n] = (__bf16)g;
                } else { // EP_BIAS_RESID
                    outF[(size_t)m * N + n] = resid[(size_t)m * N + n] + bias[n] + c;
                }
            }
        }
    }
}

// ---------------------------------------------------------------------------
// Flash attention, one block per (128 queries, head).  8 waves x 16 queries.
// Q,K fragments loaded straight from global as contiguous 16-k halves; V goes
// through an LDS transpose; P re-fragmented via per-wave LDS patch.
// ---------------------------------------------------------------------------
__global__ __launch_bounds__(256)
void attn_fa(const __bf16* __restrict__ Q, const __bf16* __restrict__ Kb,
             const __bf16* __restrict__ Vb, __bf16* __restrict__ O) {
    const int bh = blockIdx.y;                 // b*NHEAD + h
    const int b = bh >> 4, h = bh & 15;
    const int q0 = blockIdx.x * 128;
    const int tid = threadIdx.x, lane = tid & 31, wid = tid >> 5;
    const int fr = lane & 15, halfSel = lane >> 4;
    const size_t headBase = (size_t)bh * SEQ * DHEAD;

    __shared__ __attribute__((aligned(32))) __bf16 Vt[64 * 80];       // V^T tile
    __shared__ __attribute__((aligned(32))) __bf16 Pl[8][16 * 80];    // per-wave P

    const int qrow = q0 + wid * 16 + fr;
    v16bf qf[2];
    #pragma unroll
    for (int s = 0; s < 2; ++s)
        qf[s] = *(const v16bf*)&Q[headBase + (size_t)qrow * DHEAD + s * 32 + halfSel * 16];

    const v8f vzero = {0.f,0.f,0.f,0.f,0.f,0.f,0.f,0.f};
    v8f oacc[4];
    #pragma unroll
    for (int ni = 0; ni < 4; ++ni) oacc[ni] = vzero;
    float mrun[8], lrun[8];
    #pragma unroll
    for (int r = 0; r < 8; ++r) { mrun[r] = -1e30f; lrun[r] = 0.f; }

    const float scale = 0.125f;   // 1/sqrt(64)

    for (int j0 = 0; j0 < SEQ; j0 += 64) {
        __syncthreads();
        {   // cooperative transpose of V block into LDS: Vt[d][j]
            const int j = tid >> 2;      // 0..63
            const int c = tid & 3;       // 16-wide d-chunk
            v16bf vv = *(const v16bf*)&Vb[headBase + (size_t)(j0 + j) * DHEAD + c * 16];
            #pragma unroll
            for (int i = 0; i < 16; ++i) Vt[(c * 16 + i) * 80 + j] = vv[i];
        }
        __syncthreads();

        // ---- S = Q K^T  (16 x 64 per wave) ----
        v8f sacc[4];
        #pragma unroll
        for (int ni = 0; ni < 4; ++ni) sacc[ni] = vzero;
        #pragma unroll
        for (int ni = 0; ni < 4; ++ni)
            #pragma unroll
            for (int s = 0; s < 2; ++s) {
                v16bf kf = *(const v16bf*)&Kb[headBase +
                    (size_t)(j0 + ni * 16 + fr) * DHEAD + s * 32 + halfSel * 16];
                sacc[ni] = __builtin_amdgcn_wmma_f32_16x16x32_bf16(
                    false, qf[s], false, kf, (short)0, sacc[ni], false, false);
            }

        // ---- online softmax (per-row stats; reductions within 16-lane half) ----
        float pnew[4][8];
        #pragma unroll
        for (int r = 0; r < 8; ++r) {
            float mx = -1e30f;
            #pragma unroll
            for (int ni = 0; ni < 4; ++ni) mx = fmaxf(mx, sacc[ni][r] * scale);
            #pragma unroll
            for (int msk = 1; msk < 16; msk <<= 1) mx = fmaxf(mx, __shfl_xor(mx, msk, 32));
            const float mnew = fmaxf(mrun[r], mx);
            const float alpha = __expf(mrun[r] - mnew);
            float rs = 0.f;
            #pragma unroll
            for (int ni = 0; ni < 4; ++ni) {
                const float p = __expf(sacc[ni][r] * scale - mnew);
                pnew[ni][r] = p;
                rs += p;
            }
            #pragma unroll
            for (int msk = 1; msk < 16; msk <<= 1) rs += __shfl_xor(rs, msk, 32);
            lrun[r] = lrun[r] * alpha + rs;
            mrun[r] = mnew;
            #pragma unroll
            for (int ni = 0; ni < 4; ++ni) oacc[ni][r] *= alpha;
        }

        // ---- stage P (bf16) in per-wave LDS patch, re-fragment, O += P V ----
        __bf16* pw = &Pl[wid][0];
        #pragma unroll
        for (int ni = 0; ni < 4; ++ni)
            #pragma unroll
            for (int r = 0; r < 8; ++r)
                pw[(halfSel * 8 + r) * 80 + ni * 16 + fr] = (__bf16)pnew[ni][r];

        #pragma unroll
        for (int s = 0; s < 2; ++s) {
            v16bf pf = *(const v16bf*)&pw[fr * 80 + s * 32 + halfSel * 16];
            #pragma unroll
            for (int ni = 0; ni < 4; ++ni) {
                v16bf vf = *(const v16bf*)&Vt[(ni * 16 + fr) * 80 + s * 32 + halfSel * 16];
                oacc[ni] = __builtin_amdgcn_wmma_f32_16x16x32_bf16(
                    false, pf, false, vf, (short)0, oacc[ni], false, false);
            }
        }
    }

    #pragma unroll
    for (int r = 0; r < 8; ++r) {
        const float inv = 1.f / lrun[r];
        const int qOut = q0 + wid * 16 + halfSel * 8 + r;
        #pragma unroll
        for (int ni = 0; ni < 4; ++ni)
            O[((size_t)(b * SEQ + qOut)) * D_MODEL + h * DHEAD + ni * 16 + fr] =
                (__bf16)(oacc[ni][r] * inv);
    }
}

// ---------------------------------------------------------------------------
// host orchestration
// ---------------------------------------------------------------------------
extern "C" void kernel_launch(void* const* d_in, const int* in_sizes, int n_in,
                              void* d_out, int out_size, void* d_ws, size_t ws_size,
                              hipStream_t stream) {
    (void)in_sizes; (void)n_in; (void)out_size; (void)ws_size;
    const float* x      = (const float*)d_in[0];
    const float* ln1s   = (const float*)d_in[1];
    const float* ln1b   = (const float*)d_in[2];
    const float* qkvw_f = (const float*)d_in[3];
    const float* outw_f = (const float*)d_in[4];
    const float* ln2s   = (const float*)d_in[5];
    const float* ln2b   = (const float*)d_in[6];
    const float* fc1w_f = (const float*)d_in[7];
    const float* fc1b   = (const float*)d_in[8];
    const float* fc2w_f = (const float*)d_in[9];
    const float* fc2b   = (const float*)d_in[10];
    float* out = (float*)d_out;

    char* ws = (char*)d_ws;
    size_t off = 0;
    auto alloc = [&](size_t bytes) -> void* {
        void* p = ws + off;
        off += (bytes + 255) & ~(size_t)255;
        return p;
    };
    __bf16* qkvw = (__bf16*)alloc((size_t)3 * D_MODEL * D_MODEL * 2);   //  6 MB
    __bf16* outw = (__bf16*)alloc((size_t)D_MODEL * D_MODEL * 2);       //  2 MB
    __bf16* fc1w = (__bf16*)alloc((size_t)DIM_FF * D_MODEL * 2);        //  8 MB
    __bf16* fc2w = (__bf16*)alloc((size_t)D_MODEL * DIM_FF * 2);        //  8 MB
    __bf16* hbf  = (__bf16*)alloc((size_t)MROWS * D_MODEL * 2);         // 16 MB (LN1; reused for LN2)
    __bf16* Qb   = (__bf16*)alloc((size_t)MROWS * D_MODEL * 2);         // 16 MB
    __bf16* Kb   = (__bf16*)alloc((size_t)MROWS * D_MODEL * 2);         // 16 MB
    __bf16* Vb   = (__bf16*)alloc((size_t)MROWS * D_MODEL * 2);         // 16 MB
    __bf16* Obf  = (__bf16*)alloc((size_t)MROWS * D_MODEL * 2);         // 16 MB
    float*  x1   = (float*)alloc((size_t)MROWS * D_MODEL * 4);          // 32 MB
    __bf16* a1   = Qb;    // FFN act (64 MB) reuses dead Q|K|V|O span
    __bf16* h2bf = hbf;   // LN2 output reuses dead LN1 buffer

    // 1) weights -> bf16
    cvt_f32_bf16<<<(3 * D_MODEL * D_MODEL) / 1024, 256, 0, stream>>>(qkvw_f, qkvw, 3 * D_MODEL * D_MODEL);
    cvt_f32_bf16<<<(D_MODEL * D_MODEL) / 1024, 256, 0, stream>>>(outw_f, outw, D_MODEL * D_MODEL);
    cvt_f32_bf16<<<(DIM_FF * D_MODEL) / 1024, 256, 0, stream>>>(fc1w_f, fc1w, DIM_FF * D_MODEL);
    cvt_f32_bf16<<<(D_MODEL * DIM_FF) / 1024, 256, 0, stream>>>(fc2w_f, fc2w, D_MODEL * DIM_FF);

    // 2) LN1
    ln_to_bf16<<<MROWS, 256, 0, stream>>>(x, ln1s, ln1b, hbf);

    // 3) QKV projection: [8192,1024] x [3072,1024]^T, scatter into heads
    gemm_bf16<EP_QKV><<<dim3(3 * D_MODEL / 128, MROWS / 128), 256, 0, stream>>>(
        hbf, qkvw, MROWS, 3 * D_MODEL, D_MODEL,
        nullptr, nullptr, nullptr, nullptr, Qb, Kb, Vb);

    // 4) flash attention -> Obf [B,T,D]
    attn_fa<<<dim3(SEQ / 128, BATCH * NHEAD), 256, 0, stream>>>(Qb, Kb, Vb, Obf);

    // 5) out projection + residual -> x1 (fp32)
    gemm_bf16<EP_RESID><<<dim3(D_MODEL / 128, MROWS / 128), 256, 0, stream>>>(
        Obf, outw, MROWS, D_MODEL, D_MODEL,
        x, nullptr, x1, nullptr, nullptr, nullptr, nullptr);

    // 6) LN2
    ln_to_bf16<<<MROWS, 256, 0, stream>>>(x1, ln2s, ln2b, h2bf);

    // 7) fc1 + bias + exact GELU -> a1 (bf16)
    gemm_bf16<EP_BIAS_GELU><<<dim3(DIM_FF / 128, MROWS / 128), 256, 0, stream>>>(
        h2bf, fc1w, MROWS, DIM_FF, D_MODEL,
        nullptr, fc1b, nullptr, a1, nullptr, nullptr, nullptr);

    // 8) fc2 + bias + residual -> d_out (fp32)
    gemm_bf16<EP_BIAS_RESID><<<dim3(D_MODEL / 128, MROWS / 128), 256, 0, stream>>>(
        a1, fc2w, MROWS, D_MODEL, DIM_FF,
        x1, fc2b, out, nullptr, nullptr, nullptr, nullptr);
}
